// GCN_78546361909531
// MI455X (gfx1250) — compile-verified, matrix operand order checked by reading, not run on hardware
//
#include <hip/hip_runtime.h>
#include <hip/hip_bf16.h>

#define N_NODES   10000
#define NUM_GENES 2000
#define EMBED_DIM 301
#define XW_STRIDE 304            // padded row stride for xw scratch
#define NUM_EDGES 320000
#define NT_PAD    20             // padded N tiles (covers 301 -> 320)
#define KCHUNKS   63             // ceil(2000/32)
#define KFULL     62             // full 32-wide chunks (62*32 = 1984)
#define BPK_ELEMS (NT_PAD * KCHUNKS * 32 * 16)   // 645,120 halves per plane
#define OUT_TILE_ELEMS (10000ULL * 4ULL * 301ULL * 12ULL)   // 144,480,000

typedef __attribute__((ext_vector_type(16))) __bf16 v16bf;
typedef __attribute__((ext_vector_type(8)))  float  v8f;
typedef __attribute__((ext_vector_type(4)))  float  v4f;

// ---- bf16 helpers (RNE via bit trick; inputs are finite) ----
__device__ __forceinline__ float bf2f(unsigned short h) {
  unsigned u = ((unsigned)h) << 16; float f; __builtin_memcpy(&f, &u, 4); return f;
}
__device__ __forceinline__ unsigned short f2bf(float f) {
  unsigned u; __builtin_memcpy(&u, &f, 4);
  u += 0x7FFFu + ((u >> 16) & 1u);
  return (unsigned short)(u >> 16);
}
__device__ __forceinline__ __bf16 us2bf(unsigned short h) {
  __bf16 b; __builtin_memcpy(&b, &h, 2); return b;
}
// split 4 floats into hi/lo bf16 halves [base, base+3] of a fragment
__device__ __forceinline__ void cvt4(v4f q, v16bf& hi, v16bf& lo, int base) {
#pragma unroll
  for (int j = 0; j < 4; ++j) {
    float v = q[j];
    unsigned short hb = f2bf(v);
    float lf = v - bf2f(hb);
    hi[base + j] = us2bf(hb);
    lo[base + j] = us2bf(f2bf(lf));
  }
}

// ---- 0) zero deg + agg scratch (runs every call: graph-replay safe) ----
__global__ __launch_bounds__(256) void zero_kernel(float* __restrict__ deg,
                                                   float* __restrict__ agg) {
  int t = blockIdx.x * blockDim.x + threadIdx.x;
  if (t < N_NODES) deg[t] = 0.0f;
  if (t < N_NODES * EMBED_DIM) agg[t] = 0.0f;
}

// ---- 1) pack W into WMMA B-fragment layout, bf16 hi/lo planes, zero-padded ----
// index t = ((nt*63 + c)*32 + lane)*16 + h ; value = W[c*32 + kmap(h,g)][nt*16 + nl]
__global__ __launch_bounds__(256) void packB_kernel(const float* __restrict__ Wm,
                                                    __bf16* __restrict__ bhi,
                                                    __bf16* __restrict__ blo) {
  int t = blockIdx.x * blockDim.x + threadIdx.x;
  if (t >= BPK_ELEMS) return;
  int h    = t & 15;
  int lane = (t >> 4) & 31;
  int c    = (t >> 9) % KCHUNKS;
  int nt   = (t >> 9) / KCHUNKS;
  int g    = lane >> 4;
  int k    = c * 32 + h + ((h < 8) ? 0 : 8) + (g << 3);
  int n    = nt * 16 + (lane & 15);
  float v = (k < NUM_GENES && n < EMBED_DIM) ? Wm[(size_t)k * EMBED_DIM + n] : 0.0f;
  unsigned short hb = f2bf(v);
  float lf = v - bf2f(hb);
  bhi[t] = us2bf(hb);
  blo[t] = us2bf(f2bf(lf));
}

// ---- 2) degree at target nodes ----
__global__ __launch_bounds__(256) void deg_kernel(const int* __restrict__ ei,
                                                  const float* __restrict__ ew,
                                                  float* __restrict__ deg) {
  int t = blockIdx.x * blockDim.x + threadIdx.x;
  if (t < NUM_EDGES) {
    int col = ei[NUM_EDGES + t];
    if ((unsigned)col < (unsigned)N_NODES)
      unsafeAtomicAdd(&deg[col], ew[t]);
  }
}

// ---- 3) dis = (deg + selfloop)^-1/2 ----
__global__ __launch_bounds__(256) void dis_kernel(const float* __restrict__ deg,
                                                  float* __restrict__ dis) {
  int t = blockIdx.x * blockDim.x + threadIdx.x;
  if (t < N_NODES) {
    float d = deg[t] + 1.0f;
    dis[t] = (d > 0.0f) ? rsqrtf(d) : 0.0f;
  }
}

// ---- 4) xw = x @ W via bf16 hi/lo split WMMA; guard-free hot loop ----
// grid.x = 625 M-tiles; 128 threads = 4 waves; wave w owns N-tiles {w,w+4,...,w+16}
__global__ __launch_bounds__(128) void gemm_wmma(const float* __restrict__ x,
                                                 const __bf16* __restrict__ bhi,
                                                 const __bf16* __restrict__ blo,
                                                 float* __restrict__ xw) {
  const int m0   = blockIdx.x << 4;
  const int wave = threadIdx.x >> 5;
  const int lane = threadIdx.x & 31;
  const int g    = lane >> 4;       // lane group -> +8 K offset
  const int nl   = lane & 15;       // M index (A) / N index (B)

  // A row pointer, pre-offset by 8g (fragment halves 0..7 -> k0+8g+h, 8..15 -> +16)
  const float* __restrict__ xrow = x + (size_t)(m0 + nl) * NUM_GENES + (g << 3);

  // packed-B per-tile per-lane base (halves): nt*63*512 + lane*16
  size_t bbase[5];
#pragma unroll
  for (int i = 0; i < 5; ++i)
    bbase[i] = (size_t)(wave + (i << 2)) * (KCHUNKS * 512) + (size_t)lane * 16;

  v8f acc[5] = {};

  for (int c = 0; c < KFULL; ++c) {
    const v4f* p = (const v4f*)(xrow + (c << 5));   // 32B-aligned
    v4f q0 = p[0], q1 = p[1], q2 = p[4], q3 = p[5];
    v16bf a_hi, a_lo;
    cvt4(q0, a_hi, a_lo, 0);
    cvt4(q1, a_hi, a_lo, 4);
    cvt4(q2, a_hi, a_lo, 8);
    cvt4(q3, a_hi, a_lo, 12);
#pragma unroll
    for (int i = 0; i < 5; ++i) {
      const size_t off = bbase[i] + (size_t)c * 512;
      v16bf bh = *(const v16bf*)(bhi + off);
      v16bf bl = *(const v16bf*)(blo + off);
      acc[i] = __builtin_amdgcn_wmma_f32_16x16x32_bf16(
          false, a_hi, false, bh, (short)0, acc[i], false, false);
      acc[i] = __builtin_amdgcn_wmma_f32_16x16x32_bf16(
          false, a_lo, false, bh, (short)0, acc[i], false, false);
      acc[i] = __builtin_amdgcn_wmma_f32_16x16x32_bf16(
          false, a_hi, false, bl, (short)0, acc[i], false, false);
    }
  }

  // K tail: chunk 62 covers k 1984..2015; only halves 0..7 (k 1984+8g+0..7) valid.
  {
    const int c = KFULL;
    const v4f* p = (const v4f*)(xrow + (c << 5));
    v4f q0 = p[0], q1 = p[1];                    // k <= 1999: in bounds
    v16bf a_hi = {}, a_lo = {};                  // halves 8..15 stay zero
    cvt4(q0, a_hi, a_lo, 0);
    cvt4(q1, a_hi, a_lo, 4);
#pragma unroll
    for (int i = 0; i < 5; ++i) {
      const size_t off = bbase[i] + (size_t)c * 512;
      v16bf bh = *(const v16bf*)(bhi + off);
      v16bf bl = *(const v16bf*)(blo + off);
      acc[i] = __builtin_amdgcn_wmma_f32_16x16x32_bf16(
          false, a_hi, false, bh, (short)0, acc[i], false, false);
      acc[i] = __builtin_amdgcn_wmma_f32_16x16x32_bf16(
          false, a_lo, false, bh, (short)0, acc[i], false, false);
      acc[i] = __builtin_amdgcn_wmma_f32_16x16x32_bf16(
          false, a_hi, false, bl, (short)0, acc[i], false, false);
    }
  }

  // Store: C/D layout — VGPR r holds M = r + 8g, lane nl holds N
#pragma unroll
  for (int i = 0; i < 5; ++i) {
    const int n = ((wave + (i << 2)) << 4) + nl;
    if (n < EMBED_DIM) {
#pragma unroll
      for (int r = 0; r < 8; ++r) {
        const int m = m0 + r + (g << 3);
        xw[(size_t)m * XW_STRIDE + n] = acc[i][r];
      }
    }
  }
}

// ---- 5) edge aggregation: one wave per edge, lanes stride features ----
__global__ __launch_bounds__(256) void agg_kernel(const int* __restrict__ ei,
                                                  const float* __restrict__ ew,
                                                  const float* __restrict__ dis,
                                                  const float* __restrict__ xw,
                                                  float* __restrict__ agg) {
  int e    = (blockIdx.x * blockDim.x + threadIdx.x) >> 5;
  int lane = threadIdx.x & 31;
  if (e >= NUM_EDGES) return;
  int row = ei[e];
  int col = ei[NUM_EDGES + e];
  if ((unsigned)row >= (unsigned)N_NODES || (unsigned)col >= (unsigned)N_NODES) return;
  float norm = dis[row] * ew[e] * dis[col];
  const float* __restrict__ src = xw + (size_t)row * XW_STRIDE;
  float* __restrict__ dst = agg + (size_t)col * EMBED_DIM;
  for (int d = lane; d < EMBED_DIM; d += 32)
    unsafeAtomicAdd(&dst[d], src[d] * norm);
}

// ---- 6) self-loop + bias + relu; write embeddings and 4x12 broadcast (NT stores) ----
__global__ __launch_bounds__(256) void finalize_kernel(const float* __restrict__ agg,
                                                       const float* __restrict__ dis,
                                                       const float* __restrict__ xw,
                                                       const float* __restrict__ b,
                                                       float* __restrict__ out) {
  int t = blockIdx.x * blockDim.x + threadIdx.x;   // over N*EMBED_DIM
  if (t >= N_NODES * EMBED_DIM) return;
  int n = t / EMBED_DIM;
  int d = t - n * EMBED_DIM;
  float sn = dis[n];
  float h = agg[t] + sn * sn * xw[(size_t)n * XW_STRIDE + d] + b[d];
  h = fmaxf(h, 0.0f);
  out[OUT_TILE_ELEMS + (size_t)t] = h;             // embeddings (2nd output)
  v4f v4 = {h, h, h, h};
#pragma unroll
  for (int hh = 0; hh < 4; ++hh) {
    v4f* p = (v4f*)(out + ((size_t)(n * 4 + hh) * EMBED_DIM + d) * 12ULL);
    __builtin_nontemporal_store(v4, p + 0);        // 12 floats, 48B, 16B-aligned
    __builtin_nontemporal_store(v4, p + 1);
    __builtin_nontemporal_store(v4, p + 2);
  }
}

extern "C" void kernel_launch(void* const* d_in, const int* in_sizes, int n_in,
                              void* d_out, int out_size, void* d_ws, size_t ws_size,
                              hipStream_t stream) {
  const float* x  = (const float*)d_in[0];
  const int*   ei = (const int*)d_in[1];
  const float* ew = (const float*)d_in[2];
  // d_in[3] = coordinate (unused by the reference outputs)
  const float* Wm = (const float*)d_in[4];
  const float* b  = (const float*)d_in[5];
  float* out = (float*)d_out;

  float* f   = (float*)d_ws;
  float* deg = f;                                        // 10,000 f
  float* dis = f + 16384;                                // 10,000 f
  float* xw  = f + 32768;                                // 10,000*304 f
  float* agg = f + 32768 + (size_t)N_NODES * XW_STRIDE;  // 10,000*301 f
  __bf16* bhi = (__bf16*)(agg + (size_t)N_NODES * EMBED_DIM);   // 645,120 bf16
  __bf16* blo = bhi + BPK_ELEMS;                                // 645,120 bf16

  zero_kernel<<<(N_NODES * EMBED_DIM + 255) / 256, 256, 0, stream>>>(deg, agg);
  packB_kernel<<<(BPK_ELEMS + 255) / 256, 256, 0, stream>>>(Wm, bhi, blo);
  deg_kernel<<<(NUM_EDGES + 255) / 256, 256, 0, stream>>>(ei, ew, deg);
  dis_kernel<<<(N_NODES + 255) / 256, 256, 0, stream>>>(deg, dis);
  gemm_wmma<<<N_NODES / 16, 128, 0, stream>>>(x, bhi, blo, xw);
  agg_kernel<<<(NUM_EDGES * 32 + 255) / 256, 256, 0, stream>>>(ei, ew, dis, xw, agg);
  finalize_kernel<<<(N_NODES * EMBED_DIM + 255) / 256, 256, 0, stream>>>(agg, dis, xw, b, out);
}
